// NTBXentLoss_20804821582530
// MI455X (gfx1250) — matrix operand order repaired and checked by reading, not compiled, not run on hardware
//
#include <hip/hip_runtime.h>
#include <hip/hip_bf16.h>
#include <math.h>

typedef __attribute__((ext_vector_type(16))) __bf16 v16bf;
typedef __attribute__((ext_vector_type(8)))  float  v8f;

#define B_ROWS 2048
#define N_ROWS 4096
#define DIM    1024
#define MT     128                       // N_ROWS / 32 macro-tile rows
#define TOTAL_MTILES (MT*(MT+1)/2)       // 8256 upper-tri (incl. diagonal) 32x32 tiles

// ---------------- kernel 0: zero the two f64 accumulators ----------------
__global__ void init_acc_kernel(double* acc) {
    if (threadIdx.x < 2) acc[threadIdx.x] = 0.0;
}

// ---------------- kernel 1: row-normalize and convert to bf16 ----------------
// One 256-thread block per row; each thread handles one float4 (256*4 = 1024).
__global__ void normalize_kernel(const float* __restrict__ stereos,
                                 const float* __restrict__ astereos,
                                 __hip_bfloat16* __restrict__ xn) {
    const int row = blockIdx.x;
    const float* src = (row < B_ROWS) ? (stereos + (size_t)row * DIM)
                                      : (astereos + (size_t)(row - B_ROWS) * DIM);
    const int t = threadIdx.x;
    float4 v = ((const float4*)src)[t];
    float ss = v.x*v.x + v.y*v.y + v.z*v.z + v.w*v.w;

    #pragma unroll
    for (int off = 16; off > 0; off >>= 1)
        ss += __shfl_xor(ss, off, 32);
    __shared__ float wsum[8];
    if ((t & 31) == 0) wsum[t >> 5] = ss;
    __syncthreads();
    float total = 0.f;
    #pragma unroll
    for (int w = 0; w < 8; ++w) total += wsum[w];

    const float scale = 1.0f / fmaxf(sqrtf(total), 1e-8f);  // x / max(||x||, eps)
    __hip_bfloat16* dst = xn + (size_t)row * DIM + (size_t)t * 4;
    dst[0] = __float2bfloat16(v.x * scale);
    dst[1] = __float2bfloat16(v.y * scale);
    dst[2] = __float2bfloat16(v.z * scale);
    dst[3] = __float2bfloat16(v.w * scale);
}

// ---------------- kernel 2: fused symmetric GEMM + BCE + reduction ----------------
// 8 waves per block; each wave computes a 32x32 macro-tile of csm = xn @ xn^T as a
// 2x2 grid of v_wmma_f32_16x16x32_bf16 tiles (register-blocked: each A/B fragment
// feeds two WMMAs -> 32 B/lane per WMMA from L2), then applies the BCE epilogue
// in-register and reduces to two f64 accumulators.
__global__ void __launch_bounds__(256)
tile_bce_kernel(const __hip_bfloat16* __restrict__ xn, double* __restrict__ acc) {
    const unsigned lane = threadIdx.x & 31u;
    const unsigned wave = threadIdx.x >> 5;
    const unsigned tile = blockIdx.x * 8u + wave;
    if (tile >= (unsigned)TOTAL_MTILES) return;

    // Decode linear tile -> (I, J) with I <= J over MT rows. f(I) = I*MT - I*(I-1)/2.
    int I = (int)((2.0 * MT + 1.0
                   - sqrt((2.0 * MT + 1.0) * (2.0 * MT + 1.0) - 8.0 * (double)tile))
                  * 0.5);
    if (I < 0) I = 0;
    if (I > MT - 1) I = MT - 1;
    while (I > 0 && (unsigned)(I * MT - (I * (I - 1)) / 2) > tile) --I;
    while ((unsigned)((I + 1) * MT - ((I + 1) * I) / 2) <= tile) ++I;
    const int J = I + (int)(tile - (unsigned)(I * MT - (I * (I - 1)) / 2));

    const int i0 = I * 32, j0 = J * 32;
    const unsigned m    = lane & 15u;   // row within A-subtile / col within B-subtile
    const unsigned half = lane >> 4;    // K-group selector per ISA layout

    // xn is row-major [4096][1024] bf16 -> 128 uint4 (8 bf16 each) per row.
    const uint4* xq = (const uint4*)xn;
    const unsigned arow0 = (unsigned)(i0      + (int)m) * 128u;
    const unsigned arow1 = (unsigned)(i0 + 16 + (int)m) * 128u;
    const unsigned brow0 = (unsigned)(j0      + (int)m) * 128u;
    const unsigned brow1 = (unsigned)(j0 + 16 + (int)m) * 128u;

    v8f c00 = {}, c01 = {}, c10 = {}, c11 = {};
    #pragma unroll 2
    for (int k8 = 0; k8 < 128; k8 += 4) {     // K step of 32 bf16 = 4 uint4
        union { uint4 u[2]; v16bf v; } A0, A1, B0, B1;
        // A 16x32: lane holds row, K chunks [half*8,+8) and [16+half*8,+8)
        A0.u[0] = xq[arow0 + (unsigned)k8 + half];
        A0.u[1] = xq[arow0 + (unsigned)k8 + half + 2u];
        A1.u[0] = xq[arow1 + (unsigned)k8 + half];
        A1.u[1] = xq[arow1 + (unsigned)k8 + half + 2u];
        // B 32x16: lane holds column, K chunks [half*8,+8) and [16+half*8,+8)
        B0.u[0] = xq[brow0 + (unsigned)k8 + half];
        B0.u[1] = xq[brow0 + (unsigned)k8 + half + 2u];
        B1.u[0] = xq[brow1 + (unsigned)k8 + half];
        B1.u[1] = xq[brow1 + (unsigned)k8 + half + 2u];
        c00 = __builtin_amdgcn_wmma_f32_16x16x32_bf16(false, A0.v, false, B0.v,
                                                      (short)0, c00, false, false);
        c01 = __builtin_amdgcn_wmma_f32_16x16x32_bf16(false, A0.v, false, B1.v,
                                                      (short)0, c01, false, false);
        c10 = __builtin_amdgcn_wmma_f32_16x16x32_bf16(false, A1.v, false, B0.v,
                                                      (short)0, c10, false, false);
        c11 = __builtin_amdgcn_wmma_f32_16x16x32_bf16(false, A1.v, false, B1.v,
                                                      (short)0, c11, false, false);
    }

    // Epilogue: subtile (a,b), accumulator element r of lane L is
    // csm[i0 + a*16 + r + 8*half][j0 + b*16 + m].
    const bool diag = (I == J);
    const bool tpos = ((I >= MT / 2) == (J >= MT / 2));   // same-group target
    const v8f* cs[4] = { &c00, &c01, &c10, &c11 };
    float s = 0.f;
    #pragma unroll
    for (int ab = 0; ab < 4; ++ab) {
        const int a = ab >> 1, b = ab & 1;
        const int jj = j0 + b * 16 + (int)m;
        const v8f cv = *cs[ab];
        #pragma unroll
        for (int r = 0; r < 8; ++r) {
            const int   ii = i0 + a * 16 + r + 8 * (int)half;
            const float z  = 2.0f * cv[r];                // csm / TEMP, TEMP = 0.5
            float lp = -log1pf(__expf(-z));               // log sigmoid(z)
            float ln = lp - z;                            // log sigmoid(-z)
            lp = fmaxf(lp, -100.0f);
            ln = fmaxf(ln, -100.0f);
            const float bce   = tpos ? -lp : -ln;
            const bool  valid = (!diag) || (ii < jj);     // strict upper triangle only
            s += valid ? bce : 0.0f;
        }
    }
    #pragma unroll
    for (int off = 16; off > 0; off >>= 1)
        s += __shfl_xor(s, off, 32);
    if (lane == 0)
        atomicAdd(&acc[tpos ? 0 : 1], (double)s);         // global_atomic_add_f64
}

// ---------------- kernel 3: add analytic tril constant, finalize ----------------
__global__ void finalize_kernel(const double* __restrict__ acc, float* __restrict__ out) {
    const double n_pos      = 4192256.0;   // B*(B-1)
    const double n_neg      = 4194304.0;   // N*(N-1)/2 - n_pos
    const double n_tril_neg = 4198400.0;   // N*(N+1)/2 - B*(B-1), each contributing 100
    const double loss_pos = acc[0];
    const double loss_neg = acc[1] + 100.0 * n_tril_neg;
    out[0] = (float)(loss_pos / n_pos + loss_neg / n_neg);
}

extern "C" void kernel_launch(void* const* d_in, const int* in_sizes, int n_in,
                              void* d_out, int out_size, void* d_ws, size_t ws_size,
                              hipStream_t stream) {
    const float* stereos  = (const float*)d_in[0];
    const float* astereos = (const float*)d_in[1];

    double*         acc = (double*)d_ws;                                // 16 B
    __hip_bfloat16* xn  = (__hip_bfloat16*)((char*)d_ws + 256);         // 8 MB bf16 xn

    init_acc_kernel<<<1, 32, 0, stream>>>(acc);
    normalize_kernel<<<N_ROWS, 256, 0, stream>>>(stereos, astereos, xn);
    tile_bce_kernel<<<TOTAL_MTILES / 8, 256, 0, stream>>>(xn, acc);     // 1032 blocks
    finalize_kernel<<<1, 1, 0, stream>>>(acc, (float*)d_out);
}
